// TransformerStudent_81647328297279
// MI455X (gfx1250) — matrix operand order repaired and checked by reading, not compiled, not run on hardware
//
#include <hip/hip_runtime.h>

typedef __attribute__((ext_vector_type(16))) __bf16 bf16x16;
typedef __attribute__((ext_vector_type(8)))  float  f32x8;

#define B_    4
#define T_    2048
#define OBSD_ 128
#define ACT_  32
#define D_    512
#define H_    8
#define L_    4
#define DH_   64
#define DFF_  2048
#define W_    256
#define MR_   (B_ * T_)   // 8192 rows

struct U4x2 { uint4 a, b; };

// Assemble a 32-byte WMMA fragment from two 16-byte chunks (-> ds/global b128).
__device__ __forceinline__ bf16x16 frag2(uint4 x, uint4 y) {
  U4x2 p; p.a = x; p.b = y;
  return __builtin_bit_cast(bf16x16, p);
}

__device__ __forceinline__ f32x8 wmma_bf16(bf16x16 a, bf16x16 b, f32x8 c) {
  return __builtin_amdgcn_wmma_f32_16x16x32_bf16(false, a, false, b, (short)0, c,
                                                 false, false);
}

// Intra-wave LDS ordering fence (cross-lane data exchange through LDS).
__device__ __forceinline__ void wave_lds_fence() {
  asm volatile("s_wait_dscnt 0x0" ::: "memory");
}

// CDNA5 async global->LDS copy, 32 bytes per lane (2 x b128; the instruction
// offset applies to BOTH the LDS and global address).
__device__ __forceinline__ void async_copy_b256(unsigned lds_off,
                                                const void* gaddr) {
  asm volatile(
      "global_load_async_to_lds_b128 %0, %1, off\n\t"
      "global_load_async_to_lds_b128 %0, %1, off offset:16"
      :: "v"(lds_off), "v"(gaddr) : "memory");
}

__device__ __forceinline__ void wait_async0() {
  asm volatile("s_wait_asynccnt 0x0" ::: "memory");
}

// ---------------------------------------------------------------------------
// f32 -> bf16 conversion (4 elems / thread)
// ---------------------------------------------------------------------------
__global__ __launch_bounds__(256) void cvt_f32_bf16_kernel(
    const float* __restrict__ in, __bf16* __restrict__ out, long n) {
  long i = (long)blockIdx.x * 1024 + (long)threadIdx.x * 4;
  if (i + 3 < n) {
    float4 v = *(const float4*)(in + i);
    out[i + 0] = (__bf16)v.x;
    out[i + 1] = (__bf16)v.y;
    out[i + 2] = (__bf16)v.z;
    out[i + 3] = (__bf16)v.w;
  } else {
    for (int j = 0; j < 4; ++j)
      if (i + j < n) out[i + j] = (__bf16)in[i + j];
  }
}

// ---------------------------------------------------------------------------
// Tiled transpose + f32->bf16: in[K][N] f32  ->  out[N][K] bf16
// ---------------------------------------------------------------------------
__global__ __launch_bounds__(256) void transpose_cvt_kernel(
    const float* __restrict__ in, __bf16* __restrict__ out, int K, int N) {
  __shared__ float tile[32][33];
  int k0 = blockIdx.y * 32, n0 = blockIdx.x * 32;
  int tx = threadIdx.x, ty = threadIdx.y;  // 32 x 8
#pragma unroll
  for (int i = ty; i < 32; i += 8) {
    int k = k0 + i, n = n0 + tx;
    tile[i][tx] = (k < K && n < N) ? in[(size_t)k * N + n] : 0.0f;
  }
  __syncthreads();
#pragma unroll
  for (int i = ty; i < 32; i += 8) {
    int n = n0 + i, k = k0 + tx;
    if (n < N && k < K) out[(size_t)n * K + k] = (__bf16)tile[tx][i];
  }
}

// ---------------------------------------------------------------------------
// Repack V transposed per (batch, head): vT[b][h][d][t] = V[b,t,h,d]
// ---------------------------------------------------------------------------
__global__ __launch_bounds__(256) void v_repack_kernel(
    const __bf16* __restrict__ qkv, __bf16* __restrict__ vT) {
  size_t idx = (size_t)blockIdx.x * 256 + threadIdx.x;  // over B*H*DH*T
  int t = (int)(idx & (T_ - 1));
  size_t rest = idx >> 11;
  int d = (int)(rest & (DH_ - 1));
  int h = (int)((rest >> 6) & (H_ - 1));
  int b = (int)(rest >> 9);
  vT[idx] = qkv[((size_t)(b * T_ + t)) * (3 * D_) + 2 * D_ + h * DH_ + d];
}

// ---------------------------------------------------------------------------
// LayerNorm over D_=512, one wave per row, bf16 output
// ---------------------------------------------------------------------------
__global__ __launch_bounds__(256) void ln_kernel(
    const float* __restrict__ x, const float* __restrict__ gw,
    const float* __restrict__ gb, __bf16* __restrict__ y, int rows) {
  int wid  = (int)((blockIdx.x * 256u + threadIdx.x) >> 5);
  int lane = threadIdx.x & 31;
  if (wid >= rows) return;
  const float* xr = x + (size_t)wid * D_ + lane * 16;
  float v[16];
  *(float4*)(v + 0)  = *(const float4*)(xr + 0);
  *(float4*)(v + 4)  = *(const float4*)(xr + 4);
  *(float4*)(v + 8)  = *(const float4*)(xr + 8);
  *(float4*)(v + 12) = *(const float4*)(xr + 12);
  float s = 0.f, s2 = 0.f;
#pragma unroll
  for (int j = 0; j < 16; ++j) { s += v[j]; s2 += v[j] * v[j]; }
#pragma unroll
  for (int off = 16; off > 0; off >>= 1) {
    s  += __shfl_xor(s, off);
    s2 += __shfl_xor(s2, off);
  }
  float mean = s * (1.0f / D_);
  float var  = s2 * (1.0f / D_) - mean * mean;
  float rstd = rsqrtf(var + 1e-5f);
  float g4[16], b4[16];
  const float* gp = gw + lane * 16;
  const float* bp = gb + lane * 16;
  *(float4*)(g4 + 0) = *(const float4*)(gp + 0);
  *(float4*)(g4 + 4) = *(const float4*)(gp + 4);
  *(float4*)(g4 + 8) = *(const float4*)(gp + 8);
  *(float4*)(g4 + 12) = *(const float4*)(gp + 12);
  *(float4*)(b4 + 0) = *(const float4*)(bp + 0);
  *(float4*)(b4 + 4) = *(const float4*)(bp + 4);
  *(float4*)(b4 + 8) = *(const float4*)(bp + 8);
  *(float4*)(b4 + 12) = *(const float4*)(bp + 12);
  __bf16* yr = y + (size_t)wid * D_ + lane * 16;
#pragma unroll
  for (int j = 0; j < 16; ++j)
    yr[j] = (__bf16)((v[j] - mean) * rstd * g4[j] + b4[j]);
}

// ---------------------------------------------------------------------------
// Tiled bf16 GEMM with transposed-B operand:
//   C[M,N] = epilogue(A[M][K] * Bt[N][K]^T + bias (+resid))
// 256 threads = 8 waves; block tile 128x128; wave tile 32x64; K-step 32.
// Double-buffered LDS filled with GLOBAL_LOAD_ASYNC_TO_LDS_B128 (ASYNCcnt):
//   wait_async(0) -> barrier -> issue next tile into other buffer -> WMMA.
// M must be a multiple of 128. N guarded only when NGUARD (rows clamped).
// Residual row = row & resid_mask (mask = T-1 for pos-enc, -1 for full rows).
// ---------------------------------------------------------------------------
template <bool RELU, bool RESID, bool OUTBF, bool NGUARD>
__global__ __launch_bounds__(256) void gemm_bf16_kernel(
    const __bf16* __restrict__ A, const __bf16* __restrict__ Bt,
    const float* __restrict__ bias, const float* __restrict__ resid,
    int resid_mask, void* __restrict__ Cout, int M, int N, int K) {
  __shared__ __align__(16) __bf16 As[2][128][40];  // [row m][k], padded stride
  __shared__ __align__(16) __bf16 Bs[2][128][40];  // [row n][k], padded stride

  int tid  = threadIdx.x;
  int w    = tid >> 5;
  int lane = tid & 31;
  int m0   = blockIdx.y * 128;
  int n0   = blockIdx.x * 128;

  int wm = (w >> 1) * 32;        // wave rows base (0,32,64,96)
  int wn = (w & 1) * 64;         // wave cols base (0,64)
  int lr = lane & 15;
  int kh = (lane >> 4) * 8;      // A-frag k-half
  int kb = (lane >> 4) * 16;     // B-frag k-half

  int ldrow = tid >> 1;          // 0..127
  int ldch  = (tid & 1) * 16;    // 0 or 16

  f32x8 acc[2][4];
#pragma unroll
  for (int mi = 0; mi < 2; ++mi)
#pragma unroll
    for (int g = 0; g < 4; ++g)
#pragma unroll
      for (int r = 0; r < 8; ++r) acc[mi][g][r] = 0.0f;

  // per-thread global sources (32 B of one row per K-step)
  const __bf16* gA = A + (size_t)(m0 + ldrow) * K + ldch;
  int gn = n0 + ldrow;
  if (NGUARD && gn > N - 1) gn = N - 1;  // clamp; surplus cols discarded later
  const __bf16* gB = Bt + (size_t)gn * K + ldch;

  // per-thread LDS destinations (low 32 bits of the generic address = LDS off)
  unsigned asb[2], bsb[2];
  asb[0] = (unsigned)(size_t)&As[0][ldrow][ldch];
  asb[1] = (unsigned)(size_t)&As[1][ldrow][ldch];
  bsb[0] = (unsigned)(size_t)&Bs[0][ldrow][ldch];
  bsb[1] = (unsigned)(size_t)&Bs[1][ldrow][ldch];

  const int nsteps = K >> 5;

  // prologue: async-fill buffer 0 with K-slab 0
  async_copy_b256(asb[0], gA);
  async_copy_b256(bsb[0], gB);

  for (int s = 0; s < nsteps; ++s) {
    int cur = s & 1;
    wait_async0();     // own async writes into buf[cur] complete
    __syncthreads();   // everyone's writes complete; buf[cur^1] readers done
    if (s + 1 < nsteps) {  // async-fill other buffer; overlaps WMMA below
      int k0 = (s + 1) << 5;
      async_copy_b256(asb[cur ^ 1], gA + k0);
      async_copy_b256(bsb[cur ^ 1], gB + k0);
    }

    bf16x16 af0 = frag2(*(const uint4*)&As[cur][wm + lr][2 * kh],
                        *(const uint4*)&As[cur][wm + lr][2 * kh + 8]);
    bf16x16 af1 = frag2(*(const uint4*)&As[cur][wm + 16 + lr][2 * kh],
                        *(const uint4*)&As[cur][wm + 16 + lr][2 * kh + 8]);
#pragma unroll
    for (int g = 0; g < 4; ++g) {
      const __bf16* bp = &Bs[cur][wn + g * 16 + lr][kb];
      bf16x16 bf = frag2(*(const uint4*)bp, *(const uint4*)(bp + 8));
      acc[0][g] = wmma_bf16(af0, bf, acc[0][g]);
      acc[1][g] = wmma_bf16(af1, bf, acc[1][g]);
    }
  }

  // epilogue: C layout lane(n)=lane&15, rows m = (lane>>4)*8 + r
  int nn = lane & 15;
  int mb = (lane >> 4) * 8;
#pragma unroll
  for (int mi = 0; mi < 2; ++mi) {
#pragma unroll
    for (int g = 0; g < 4; ++g) {
      int col = n0 + wn + g * 16 + nn;
      if (NGUARD && col >= N) continue;
      float bv = bias[col];
#pragma unroll
      for (int r = 0; r < 8; ++r) {
        int row = m0 + wm + mi * 16 + mb + r;
        float v = acc[mi][g][r] + bv;
        if (RESID) v += resid[(size_t)(row & resid_mask) * N + col];
        if (RELU) v = fmaxf(v, 0.0f);
        if (OUTBF)
          ((__bf16*)Cout)[(size_t)row * N + col] = (__bf16)v;
        else
          ((float*)Cout)[(size_t)row * N + col] = v;
      }
    }
  }
}

// ---------------------------------------------------------------------------
// Sliding-window causal flash attention.
// One wave per (batch, head, 16-query tile); 32-key tiles, online softmax.
// qkv: [B*T,1536] bf16 (q|k|v); vT: [B][H][DH][T] bf16; out: [B*T,512] bf16.
// Key tiles are 32-aligned inside [0,T) -> no index clamping needed.
// Interior tiles (kt+31<=q0 && kt>=q0-240) skip all mask compares (uniform
// scalar branch); edge tiles use one unsigned range compare per element.
// ---------------------------------------------------------------------------
__global__ __launch_bounds__(128) void attn_kernel(
    const __bf16* __restrict__ qkv, const __bf16* __restrict__ vT,
    __bf16* __restrict__ attn_out) {
  __shared__ __align__(16) float  St[4][16][36];
  __shared__ __align__(16) __bf16 Pt[4][16][32];
  __shared__ float rmaxS[4][16];
  __shared__ float rsumS[4][16];
  __shared__ float alphaS[4][16];

  int w    = threadIdx.x >> 5;
  int lane = threadIdx.x & 31;
  int gw   = blockIdx.x * 4 + w;
  int qt   = gw & 127;
  int h    = (gw >> 7) & 7;
  int bidx = gw >> 10;
  int q0   = qt * 16;

  const __bf16* qbase = qkv + (size_t)(bidx * T_) * (3 * D_) + h * DH_;
  const __bf16* kbase = qbase + D_;
  const __bf16* vtb   = vT + ((size_t)(bidx * H_ + h)) * DH_ * T_;

  if (lane < 16) {
    rmaxS[w][lane] = -1e9f;
    rsumS[w][lane] = 0.0f;
  }

  int arow = lane & 15;
  int kh   = (lane >> 4) * 8;
  int nn   = lane & 15;
  int mb   = (lane >> 4) * 8;
  int kb   = (lane >> 4) * 16;

  // Q A-fragments for head-dim halves [0,32) and [32,64): 16B chunk loads
  const __bf16* qrow = qbase + (size_t)(q0 + arow) * (3 * D_);
  bf16x16 aq0 = frag2(*(const uint4*)(qrow + 2 * kh),
                      *(const uint4*)(qrow + 2 * kh + 8));
  bf16x16 aq1 = frag2(*(const uint4*)(qrow + 32 + 2 * kh),
                      *(const uint4*)(qrow + 32 + 2 * kh + 8));

  f32x8 acc[4];
#pragma unroll
  for (int g = 0; g < 4; ++g)
#pragma unroll
    for (int r = 0; r < 8; ++r) acc[g][r] = 0.0f;

  int kstart = q0 - (W_ - 1);
  if (kstart < 0) kstart = 0;
  kstart &= ~31;

  for (int kt = kstart; kt <= q0 + 15; kt += 32) {
    // uniform (scalar) tile classification: fully-inside-band tiles need no
    // per-element mask at all.
    bool full = (kt + 31 <= q0) && (kt >= q0 - (W_ - 16));

    // ---- scores: S[16q x 32k] as two 16x16 WMMA tiles, K-dim = DH (2 steps)
#pragma unroll
    for (int g = 0; g < 2; ++g) {
      const __bf16* krow = kbase + (size_t)(kt + g * 16 + nn) * (3 * D_);
      bf16x16 bk0 = frag2(*(const uint4*)(krow + kb),
                          *(const uint4*)(krow + kb + 8));
      bf16x16 bk1 = frag2(*(const uint4*)(krow + 32 + kb),
                          *(const uint4*)(krow + 32 + kb + 8));
      f32x8 s;
#pragma unroll
      for (int r = 0; r < 8; ++r) s[r] = 0.0f;
      s = wmma_bf16(aq0, bk0, s);
      s = wmma_bf16(aq1, bk1, s);
      if (full) {
#pragma unroll
        for (int r = 0; r < 8; ++r)
          St[w][mb + r][g * 16 + nn] = s[r] * 0.125f;
      } else {
        int dbase = (q0 + mb) - (kt + g * 16 + nn);  // qg - kg at r=0
#pragma unroll
        for (int r = 0; r < 8; ++r) {
          bool ok = ((unsigned)(dbase + r)) < (unsigned)W_;
          St[w][mb + r][g * 16 + nn] = ok ? s[r] * 0.125f : -1e9f;
        }
      }
    }
    wave_lds_fence();

    // ---- online softmax: 2 lanes per row (16 cols each), vector LDS reads
    {
      int row  = lane >> 1;
      int half = lane & 1;
      float sv[16];
      *(float4*)(sv + 0)  = *(const float4*)&St[w][row][half * 16 + 0];
      *(float4*)(sv + 4)  = *(const float4*)&St[w][row][half * 16 + 4];
      *(float4*)(sv + 8)  = *(const float4*)&St[w][row][half * 16 + 8];
      *(float4*)(sv + 12) = *(const float4*)&St[w][row][half * 16 + 12];
      float tmax = sv[0];
#pragma unroll
      for (int j = 1; j < 16; ++j) tmax = fmaxf(tmax, sv[j]);
      tmax = fmaxf(tmax, __shfl_xor(tmax, 1));
      float mold  = rmaxS[w][row];
      float mnew  = fmaxf(mold, tmax);
      float alpha = __expf(mold - mnew);
      float tsum  = 0.0f;
      bf16x16 pvv;
#pragma unroll
      for (int j = 0; j < 16; ++j) {
        float p = __expf(sv[j] - mnew);
        tsum += p;
        pvv[j] = (__bf16)p;
      }
      U4x2 pp = __builtin_bit_cast(U4x2, pvv);
      *(uint4*)&Pt[w][row][half * 16]     = pp.a;
      *(uint4*)&Pt[w][row][half * 16 + 8] = pp.b;
      tsum += __shfl_xor(tsum, 1);
      if (half == 0) {
        rmaxS[w][row]  = mnew;
        rsumS[w][row]  = rsumS[w][row] * alpha + tsum;
        alphaS[w][row] = alpha;
      }
    }
    wave_lds_fence();

    // ---- rescale accumulators, then O += P * V
    float al[8];
#pragma unroll
    for (int r = 0; r < 8; ++r) al[r] = alphaS[w][mb + r];
#pragma unroll
    for (int g = 0; g < 4; ++g)
#pragma unroll
      for (int r = 0; r < 8; ++r) acc[g][r] *= al[r];

    bf16x16 ap = frag2(*(const uint4*)&Pt[w][arow][2 * kh],
                       *(const uint4*)&Pt[w][arow][2 * kh + 8]);
#pragma unroll
    for (int g = 0; g < 4; ++g) {
      const __bf16* vp = vtb + (size_t)(g * 16 + nn) * T_ + kt + kb;
      bf16x16 bv = frag2(*(const uint4*)vp, *(const uint4*)(vp + 8));
      acc[g] = wmma_bf16(ap, bv, acc[g]);
    }
  }
  wave_lds_fence();

  float rs[8];
#pragma unroll
  for (int r = 0; r < 8; ++r) rs[r] = 1.0f / rsumS[w][mb + r];
#pragma unroll
  for (int g = 0; g < 4; ++g) {
    int dd = g * 16 + nn;
#pragma unroll
    for (int r = 0; r < 8; ++r) {
      int trow = q0 + mb + r;
      attn_out[(size_t)(bidx * T_ + trow) * D_ + h * DH_ + dd] =
          (__bf16)(acc[g][r] * rs[r]);
    }
  }
}

// ---------------------------------------------------------------------------
// Host orchestration
// ---------------------------------------------------------------------------
extern "C" void kernel_launch(void* const* d_in, const int* in_sizes, int n_in,
                              void* d_out, int out_size, void* d_ws,
                              size_t ws_size, hipStream_t stream) {
  (void)in_sizes; (void)n_in; (void)out_size; (void)ws_size;

  const float* obs       = (const float*)d_in[0];
  const float* pos_enc   = (const float*)d_in[1];
  const float* inp_w     = (const float*)d_in[2];
  const float* inp_b     = (const float*)d_in[3];
  const float* in_proj_w = (const float*)d_in[4];
  const float* in_proj_b = (const float*)d_in[5];
  const float* out_w     = (const float*)d_in[6];
  const float* out_b     = (const float*)d_in[7];
  const float* lin1_w    = (const float*)d_in[8];
  const float* lin1_b    = (const float*)d_in[9];
  const float* lin2_w    = (const float*)d_in[10];
  const float* lin2_b    = (const float*)d_in[11];
  const float* ln1_g     = (const float*)d_in[12];
  const float* ln1_bb    = (const float*)d_in[13];
  const float* ln2_g     = (const float*)d_in[14];
  const float* ln2_bb    = (const float*)d_in[15];
  const float* outp_w    = (const float*)d_in[16];
  const float* outp_b    = (const float*)d_in[17];

  size_t off = 0;
  auto alloc = [&](size_t bytes) -> void* {
    void* p = (char*)d_ws + off;
    off += (bytes + 255) & ~(size_t)255;
    return p;
  };

  float*  x     = (float*)alloc((size_t)MR_ * D_ * 4);
  __bf16* xb    = (__bf16*)alloc((size_t)MR_ * D_ * 2);
  __bf16* qkvb  = (__bf16*)alloc((size_t)MR_ * 3 * D_ * 2);
  __bf16* attnb = (__bf16*)alloc((size_t)MR_ * D_ * 2);
  __bf16* ffnb  = (__bf16*)alloc((size_t)MR_ * DFF_ * 2);
  __bf16* vTb   = (__bf16*)alloc((size_t)B_ * H_ * DH_ * T_ * 2);

  __bf16* obs_bf = (__bf16*)alloc((size_t)B_ * T_ * OBSD_ * 2);
  __bf16* inp_wT = (__bf16*)alloc((size_t)D_ * OBSD_ * 2);
  __bf16* inprjT = (__bf16*)alloc((size_t)L_ * 3 * D_ * D_ * 2);
  __bf16* out_wT = (__bf16*)alloc((size_t)L_ * D_ * D_ * 2);
  __bf16* lin1T  = (__bf16*)alloc((size_t)L_ * DFF_ * D_ * 2);
  __bf16* lin2T  = (__bf16*)alloc((size_t)L_ * D_ * DFF_ * 2);
  __bf16* outpT  = (__bf16*)alloc((size_t)ACT_ * D_ * 2);

  auto cvt = [&](const float* src, __bf16* dst, size_t n) {
    dim3 g((unsigned)((n + 1023) / 1024));
    cvt_f32_bf16_kernel<<<g, 256, 0, stream>>>(src, dst, (long)n);
  };
  auto tr = [&](const float* src, __bf16* dst, int K, int N) {
    dim3 g((unsigned)((N + 31) / 32), (unsigned)((K + 31) / 32));
    transpose_cvt_kernel<<<g, dim3(32, 8), 0, stream>>>(src, dst, K, N);
  };
  auto gdim = [](int M, int N) {
    return dim3((unsigned)((N + 127) / 128), (unsigned)((M + 127) / 128));
  };

  // one-time (per call) conversions: A operands plain bf16, weights transposed
  cvt(obs, obs_bf, (size_t)B_ * T_ * OBSD_);
  tr(inp_w, inp_wT, OBSD_, D_);
  for (int l = 0; l < L_; ++l) {
    tr(in_proj_w + (size_t)l * D_ * 3 * D_, inprjT + (size_t)l * 3 * D_ * D_,
       D_, 3 * D_);
    tr(out_w + (size_t)l * D_ * D_, out_wT + (size_t)l * D_ * D_, D_, D_);
    tr(lin1_w + (size_t)l * D_ * DFF_, lin1T + (size_t)l * DFF_ * D_, D_, DFF_);
    tr(lin2_w + (size_t)l * DFF_ * D_, lin2T + (size_t)l * D_ * DFF_, DFF_, D_);
  }
  tr(outp_w, outpT, D_, ACT_);

  // x = obs @ inp_w + inp_b + pos_enc   (pos_enc as row-masked residual)
  gemm_bf16_kernel<false, true, false, false>
      <<<gdim(MR_, D_), 256, 0, stream>>>(obs_bf, inp_wT, inp_b, pos_enc,
                                          T_ - 1, (void*)x, MR_, D_, OBSD_);

  for (int l = 0; l < L_; ++l) {
    ln_kernel<<<dim3(MR_ / 8), 256, 0, stream>>>(x, ln1_g + l * D_,
                                                 ln1_bb + l * D_, xb, MR_);
    gemm_bf16_kernel<false, false, true, false>
        <<<gdim(MR_, 3 * D_), 256, 0, stream>>>(
            xb, inprjT + (size_t)l * 3 * D_ * D_, in_proj_b + l * 3 * D_,
            nullptr, 0, (void*)qkvb, MR_, 3 * D_, D_);
    v_repack_kernel<<<dim3((B_ * H_ * DH_ * T_) / 256), 256, 0, stream>>>(qkvb,
                                                                          vTb);
    attn_kernel<<<dim3(B_ * H_ * (T_ / 16) / 4), 128, 0, stream>>>(qkvb, vTb,
                                                                   attnb);
    gemm_bf16_kernel<false, true, false, false>
        <<<gdim(MR_, D_), 256, 0, stream>>>(
            attnb, out_wT + (size_t)l * D_ * D_, out_b + l * D_, x, -1,
            (void*)x, MR_, D_, D_);
    ln_kernel<<<dim3(MR_ / 8), 256, 0, stream>>>(x, ln2_g + l * D_,
                                                 ln2_bb + l * D_, xb, MR_);
    gemm_bf16_kernel<true, false, true, false>
        <<<gdim(MR_, DFF_), 256, 0, stream>>>(
            xb, lin1T + (size_t)l * DFF_ * D_, lin1_b + l * DFF_, nullptr, 0,
            (void*)ffnb, MR_, DFF_, D_);
    gemm_bf16_kernel<false, true, false, false>
        <<<gdim(MR_, D_), 256, 0, stream>>>(
            ffnb, lin2T + (size_t)l * D_ * DFF_, lin2_b + l * D_, x, -1,
            (void*)x, MR_, D_, DFF_);
  }

  // out = x @ outp_w + outp_b  (N = 32, column-guarded)
  cvt(x, xb, (size_t)MR_ * D_);
  gemm_bf16_kernel<false, false, false, true>
      <<<gdim(MR_, ACT_), 256, 0, stream>>>(xb, outpT, outp_b, nullptr, 0,
                                            d_out, MR_, ACT_, D_);
}
